// feature_conbined_86011015070115
// MI455X (gfx1250) — matrix operand order repaired and checked by reading, not compiled, not run on hardware
//
#include <hip/hip_runtime.h>
#include <hip/hip_bf16.h>
#include <math.h>

// ---------------------------------------------------------------------------
// Sizes from the reference
// ---------------------------------------------------------------------------
#define BATCH 1024
#define LEN   256
#define FM    64      // pooled dim (256/4)
#define HEAD  64
#define EMB   48
#define KTOT  4096    // FM*FM flattened contraction dim for NDI

// K split for the big NDI GEMM: 8 parts of 512 -> 64x8 = 512 workgroups
#define KPARTS  8
#define KCHUNK  (KTOT / KPARTS)        // 512
#define ROWSTR  (KCHUNK + 8)           // half-words; pad to dodge LDS bank conflicts

typedef __attribute__((ext_vector_type(16))) _Float16 v16h;
typedef __attribute__((ext_vector_type(8)))  _Float16 h8;
typedef __attribute__((ext_vector_type(8)))  float    v8f;

union U16h { v16h v; h8 h[2]; };

__device__ __forceinline__ float elu1(float v) {
    return v > 0.0f ? v : (__expf(v) - 1.0f);
}

// ---------------------------------------------------------------------------
// Kernel 1: convert wm_ndi (HEAD x FM x FM f32) -> f16, layout [h][k] k=i*64+j
// ---------------------------------------------------------------------------
__global__ __launch_bounds__(256) void k_prep_wm(const float* __restrict__ wm,
                                                 _Float16* __restrict__ wmh) {
    int idx = blockIdx.x * 256 + threadIdx.x;           // 64*4096 = 262144
    if (idx < HEAD * KTOT) wmh[idx] = (_Float16)wm[idx];
}

// ---------------------------------------------------------------------------
// Kernel 2: effective rank-2 weights for DI and DD:
//   weff[h][k] = sum_i wm[h][i][k] - sum_j wm[h][k][j]
// ---------------------------------------------------------------------------
__global__ __launch_bounds__(256) void k_prep_weff(const float* __restrict__ wm_di,
                                                   const float* __restrict__ wm_dd,
                                                   _Float16* __restrict__ weff_di,
                                                   _Float16* __restrict__ weff_dd) {
    int idx = blockIdx.x * 256 + threadIdx.x;           // 2*64*64 = 8192
    if (idx >= 2 * HEAD * FM) return;
    int mat = idx >> 12;
    int hk  = idx & 4095;
    int h   = hk >> 6;
    int k   = hk & 63;
    const float* w = mat ? wm_dd : wm_di;
    float cs = 0.0f, rs = 0.0f;
    const float* base = w + h * KTOT;
    for (int i = 0; i < FM; ++i) {
        cs += base[i * FM + k];   // column sum over i
        rs += base[k * FM + i];   // row sum over j
    }
    _Float16* dst = mat ? weff_dd : weff_di;
    dst[hk] = (_Float16)(cs - rs);
}

// ---------------------------------------------------------------------------
// Kernel 3: per-batch smooth (5-tap edge-clamped mean), diff, pool-4 of x and
// of diff.  One block per batch row.
// ---------------------------------------------------------------------------
__global__ __launch_bounds__(256) void k_prep_x(const float* __restrict__ x,
                                                _Float16* __restrict__ xp16,
                                                _Float16* __restrict__ dp16) {
    __shared__ float xs[LEN];
    __shared__ float smv[LEN];
    __shared__ float dfv[LEN];
    int b = blockIdx.x;
    int t = threadIdx.x;
    xs[t] = x[b * LEN + t];
    __syncthreads();
    int im2 = t - 2 < 0 ? 0 : t - 2;
    int im1 = t - 1 < 0 ? 0 : t - 1;
    int ip1 = t + 1 > LEN - 1 ? LEN - 1 : t + 1;
    int ip2 = t + 2 > LEN - 1 ? LEN - 1 : t + 2;
    smv[t] = (xs[im2] + xs[im1] + xs[t] + xs[ip1] + xs[ip2]) * 0.2f;
    __syncthreads();
    dfv[t] = (t == 0) ? (smv[0] - xs[0]) : (smv[t] - smv[t - 1]);
    __syncthreads();
    if (t < FM) {
        xp16[b * FM + t] = (_Float16)((xs[4*t] + xs[4*t+1] + xs[4*t+2] + xs[4*t+3]) * 0.25f);
        dp16[b * FM + t] = (_Float16)((dfv[4*t] + dfv[4*t+1] + dfv[4*t+2] + dfv[4*t+3]) * 0.25f);
    }
}

// ---------------------------------------------------------------------------
// Kernel 4: fused NDI-panel generation + WMMA GEMMs.
// Grid = (BATCH/16, KPARTS).  blockIdx.x picks the 16-batch M-tile,
// blockIdx.y picks a 512-wide K slice.  128 threads = 4 waves, each wave owns
// an N-tile of 16 heads.  The block builds its 16x512 pooled-NDI panel in LDS
// (f16) then runs 16 v_wmma_f32_16x16x32_f16 per wave; the partial 16x64
// accumulator goes to Hp[kc] (deterministic fixed-order reduction later).
// kc==0 blocks additionally run the rank-2 DI/DD GEMMs (complete in-block)
// and store bias+ELU'd results directly.
// ---------------------------------------------------------------------------
__global__ __launch_bounds__(128) void k_gemm(const float* __restrict__ x,
                                              const _Float16* __restrict__ wmh,
                                              const _Float16* __restrict__ weff_di,
                                              const _Float16* __restrict__ weff_dd,
                                              const _Float16* __restrict__ xp16,
                                              const _Float16* __restrict__ dp16,
                                              const float* __restrict__ b_di,
                                              const float* __restrict__ b_dd,
                                              float* __restrict__ Hp,
                                              float* __restrict__ Hdi,
                                              float* __restrict__ Hdd) {
    __shared__ float    xs[16][260];            // 16 batches x 256 samples
    __shared__ _Float16 pN[16 * ROWSTR];        // pooled-NDI panel slice (f16)

    const int tid   = threadIdx.x;
    const int b0    = blockIdx.x * 16;
    const int kc    = blockIdx.y;
    const int kbase = kc * KCHUNK;
    const int lane  = tid & 31;
    const int wv    = tid >> 5;
    const int n0    = wv * 16;
    const int lm    = lane & 15;                // A row / B column within tile
    const int hi    = lane >> 4;                // lane half select

    // ---- load x tile ----
    for (int e = tid; e < 16 * LEN; e += 128) {
        int m = e >> 8, j = e & 255;
        xs[m][j] = x[(b0 + m) * LEN + j];
    }
    __syncthreads();

    // ---- build pooled-NDI panel for this K slice ----
    for (int e = tid; e < 16 * KCHUNK; e += 128) {
        int m  = e / KCHUNK;
        int kk = e % KCHUNK;
        int k  = kbase + kk;
        int i  = k >> 6, j = k & 63;
        const float* xr = xs[m];
        float av[4], cv[4];
        #pragma unroll
        for (int q = 0; q < 4; ++q) { av[q] = xr[i * 4 + q]; cv[q] = xr[j * 4 + q]; }
        float s = 0.0f;
        #pragma unroll
        for (int pi = 0; pi < 4; ++pi) {
            #pragma unroll
            for (int pj = 0; pj < 4; ++pj) {
                float num = cv[pj] - av[pi];
                float den = cv[pj] + av[pi] + 1e-5f;
                s += num * __builtin_amdgcn_rcpf(den);
            }
        }
        pN[m * ROWSTR + kk] = (_Float16)(s * 0.0625f);
    }
    __syncthreads();

    // ---- WMMA over the slice ----
    v8f c_ndi = {};
    const _Float16* Arow = &pN[lm * ROWSTR];
    const _Float16* Bcol = wmh + (size_t)(n0 + lm) * KTOT + kbase + hi * 16;
    for (int kb = 0; kb < KCHUNK; kb += 32) {
        U16h ua, ub;
        // A (16x32 f16): lane = row m, 16 consecutive K at hi*16
        ua.h[0] = *(const h8*)(Arow + kb + hi * 16);
        ua.h[1] = *(const h8*)(Arow + kb + hi * 16 + 8);
        // B (32x16 f16): lane = column n, 16 consecutive K at hi*16
        const _Float16* bp = Bcol + kb;
        __builtin_prefetch((const void*)(bp + 128), 0, 1);  // global_prefetch_b8
        ub.h[0] = *(const h8*)(bp);
        ub.h[1] = *(const h8*)(bp + 8);
        c_ndi = __builtin_amdgcn_wmma_f32_16x16x32_f16(
            false, ua.v, false, ub.v, (short)0, c_ndi, false, false);
    }

    const int n = n0 + lm;
    // partial NDI accumulator (no bias yet; reduced deterministically later)
    #pragma unroll
    for (int r = 0; r < 8; ++r) {
        int bb = b0 + r + hi * 8;
        Hp[((size_t)kc * BATCH + bb) * HEAD + n] = c_ndi[r];
    }

    if (kc == 0) {
        // ---- rank-2 DI and DD GEMMs (K = 64 -> two WMMA each) ----
        v8f c_di = {}; v8f c_dd = {};
        {
            const _Float16* Ar = xp16 + (size_t)(b0 + lm) * FM;
            const _Float16* Br = weff_di + (size_t)(n0 + lm) * FM + hi * 16;
            #pragma unroll
            for (int kb = 0; kb < FM; kb += 32) {
                U16h ua, ub;
                ua.h[0] = *(const h8*)(Ar + kb + hi * 16);
                ua.h[1] = *(const h8*)(Ar + kb + hi * 16 + 8);
                ub.h[0] = *(const h8*)(Br + kb);
                ub.h[1] = *(const h8*)(Br + kb + 8);
                c_di = __builtin_amdgcn_wmma_f32_16x16x32_f16(
                    false, ua.v, false, ub.v, (short)0, c_di, false, false);
            }
        }
        {
            const _Float16* Ar = dp16 + (size_t)(b0 + lm) * FM;
            const _Float16* Br = weff_dd + (size_t)(n0 + lm) * FM + hi * 16;
            #pragma unroll
            for (int kb = 0; kb < FM; kb += 32) {
                U16h ua, ub;
                ua.h[0] = *(const h8*)(Ar + kb + hi * 16);
                ua.h[1] = *(const h8*)(Ar + kb + hi * 16 + 8);
                ub.h[0] = *(const h8*)(Br + kb);
                ub.h[1] = *(const h8*)(Br + kb + 8);
                c_dd = __builtin_amdgcn_wmma_f32_16x16x32_f16(
                    false, ua.v, false, ub.v, (short)0, c_dd, false, false);
            }
        }
        const float biD = b_di[n], biDD = b_dd[n];
        #pragma unroll
        for (int r = 0; r < 8; ++r) {
            int bb = b0 + r + hi * 8;
            Hdi[bb * HEAD + n] = elu1(c_di[r] + biD);
            Hdd[bb * HEAD + n] = elu1(c_dd[r] + biDD);
        }
    }
}

// ---------------------------------------------------------------------------
// Kernel 5: deterministic fixed-order reduction of the KPARTS NDI partials,
// fused with bias + ELU.
// ---------------------------------------------------------------------------
__global__ __launch_bounds__(256) void k_act(const float* __restrict__ Hp,
                                             const float* __restrict__ b_ndi,
                                             float* __restrict__ Hndi) {
    int idx = blockIdx.x * 256 + threadIdx.x;      // 1024*64
    if (idx >= BATCH * HEAD) return;
    int n = idx & (HEAD - 1);
    float s = b_ndi[n];
    #pragma unroll
    for (int kc = 0; kc < KPARTS; ++kc)
        s += Hp[(size_t)kc * BATCH * HEAD + idx];
    Hndi[idx] = elu1(s);
}

// ---------------------------------------------------------------------------
// Kernel 6: per-picker FC:  E[b, p*48+o] = elu( fcw_p[o,:] . Hp[b,:] + fcb_p[o] )
// ---------------------------------------------------------------------------
__global__ __launch_bounds__(256) void k_head1(const float* __restrict__ Hdi,
                                               const float* __restrict__ Hndi,
                                               const float* __restrict__ Hdd,
                                               const float* __restrict__ fcw_di,
                                               const float* __restrict__ fcb_di,
                                               const float* __restrict__ fcw_ndi,
                                               const float* __restrict__ fcb_ndi,
                                               const float* __restrict__ fcw_dd,
                                               const float* __restrict__ fcb_dd,
                                               float* __restrict__ E) {
    int idx = blockIdx.x * 256 + threadIdx.x;       // 1024*144
    if (idx >= BATCH * 3 * EMB) return;
    int b = idx / (3 * EMB);
    int o = idx % (3 * EMB);
    int p = o / EMB;
    int oo = o % EMB;
    const float* H = (p == 0) ? Hdi : (p == 1) ? Hndi : Hdd;
    const float* W = (p == 0) ? fcw_di : (p == 1) ? fcw_ndi : fcw_dd;
    const float* Bv = (p == 0) ? fcb_di : (p == 1) ? fcb_ndi : fcb_dd;
    const float* h = H + b * HEAD;
    const float* w = W + oo * HEAD;
    float s = Bv[oo];
    for (int k = 0; k < HEAD; ++k) s += h[k] * w[k];
    E[b * (3 * EMB) + o] = elu1(s);
}

// ---------------------------------------------------------------------------
// Kernel 7: fc1 (144 -> 48) + ELU
// ---------------------------------------------------------------------------
__global__ __launch_bounds__(256) void k_head2(const float* __restrict__ E,
                                               const float* __restrict__ fc1_w,
                                               const float* __restrict__ fc1_b,
                                               float* __restrict__ M) {
    int idx = blockIdx.x * 256 + threadIdx.x;       // 1024*48
    if (idx >= BATCH * EMB) return;
    int b = idx / EMB;
    int o = idx % EMB;
    const float* e = E + b * (3 * EMB);
    const float* w = fc1_w + o * (3 * EMB);
    float s = fc1_b[o];
    for (int k = 0; k < 3 * EMB; ++k) s += e[k] * w[k];
    M[b * EMB + o] = elu1(s);
}

// ---------------------------------------------------------------------------
// Kernel 8: fc2 (48 -> 20), no activation
// ---------------------------------------------------------------------------
__global__ __launch_bounds__(256) void k_head3(const float* __restrict__ M,
                                               const float* __restrict__ fc2_w,
                                               const float* __restrict__ fc2_b,
                                               float* __restrict__ out) {
    int idx = blockIdx.x * 256 + threadIdx.x;       // 1024*20
    if (idx >= BATCH * 20) return;
    int b = idx / 20;
    int o = idx % 20;
    const float* m = M + b * EMB;
    const float* w = fc2_w + o * EMB;
    float s = fc2_b[o];
    for (int k = 0; k < EMB; ++k) s += m[k] * w[k];
    out[b * 20 + o] = s;
}

// ---------------------------------------------------------------------------
extern "C" void kernel_launch(void* const* d_in, const int* in_sizes, int n_in,
                              void* d_out, int out_size, void* d_ws, size_t ws_size,
                              hipStream_t stream) {
    (void)in_sizes; (void)n_in; (void)out_size; (void)ws_size;
    const float* x       = (const float*)d_in[0];
    const float* wm_di   = (const float*)d_in[1];
    const float* b_di    = (const float*)d_in[2];
    const float* fcw_di  = (const float*)d_in[3];
    const float* fcb_di  = (const float*)d_in[4];
    const float* wm_ndi  = (const float*)d_in[5];
    const float* b_ndi   = (const float*)d_in[6];
    const float* fcw_ndi = (const float*)d_in[7];
    const float* fcb_ndi = (const float*)d_in[8];
    const float* wm_dd   = (const float*)d_in[9];
    const float* b_dd    = (const float*)d_in[10];
    const float* fcw_dd  = (const float*)d_in[11];
    const float* fcb_dd  = (const float*)d_in[12];
    const float* fc1_w   = (const float*)d_in[13];
    const float* fc1_b   = (const float*)d_in[14];
    const float* fc2_w   = (const float*)d_in[15];
    const float* fc2_b   = (const float*)d_in[16];
    float* out = (float*)d_out;

    // workspace layout (all 16B aligned)
    char* ws = (char*)d_ws;
    size_t off = 0;
    _Float16* wmh     = (_Float16*)(ws + off); off += (size_t)HEAD * KTOT * 2;        // 512 KB
    _Float16* weff_di = (_Float16*)(ws + off); off += (size_t)HEAD * FM * 2;          // 8 KB
    _Float16* weff_dd = (_Float16*)(ws + off); off += (size_t)HEAD * FM * 2;          // 8 KB
    _Float16* xp16    = (_Float16*)(ws + off); off += (size_t)BATCH * FM * 2;         // 128 KB
    _Float16* dp16    = (_Float16*)(ws + off); off += (size_t)BATCH * FM * 2;         // 128 KB
    float*    Hp      = (float*)(ws + off);    off += (size_t)KPARTS * BATCH * HEAD * 4; // 2 MB
    float*    Hdi     = (float*)(ws + off);    off += (size_t)BATCH * HEAD * 4;       // 256 KB
    float*    Hndi    = (float*)(ws + off);    off += (size_t)BATCH * HEAD * 4;
    float*    Hdd     = (float*)(ws + off);    off += (size_t)BATCH * HEAD * 4;
    float*    E       = (float*)(ws + off);    off += (size_t)BATCH * 3 * EMB * 4;
    float*    Mt      = (float*)(ws + off);    off += (size_t)BATCH * EMB * 4;

    k_prep_wm  <<<(HEAD * KTOT + 255) / 256, 256, 0, stream>>>(wm_ndi, wmh);
    k_prep_weff<<<(2 * HEAD * FM + 255) / 256, 256, 0, stream>>>(wm_di, wm_dd, weff_di, weff_dd);
    k_prep_x   <<<BATCH, 256, 0, stream>>>(x, xp16, dp16);
    k_gemm     <<<dim3(BATCH / 16, KPARTS), 128, 0, stream>>>(
                   x, wmh, weff_di, weff_dd, xp16, dp16, b_di, b_dd, Hp, Hdi, Hdd);
    k_act      <<<(BATCH * HEAD + 255) / 256, 256, 0, stream>>>(Hp, b_ndi, Hndi);
    k_head1    <<<(BATCH * 3 * EMB + 255) / 256, 256, 0, stream>>>(
                   Hdi, Hndi, Hdd, fcw_di, fcb_di, fcw_ndi, fcb_ndi, fcw_dd, fcb_dd, E);
    k_head2    <<<(BATCH * EMB + 255) / 256, 256, 0, stream>>>(E, fc1_w, fc1_b, Mt);
    k_head3    <<<(BATCH * 20 + 255) / 256, 256, 0, stream>>>(Mt, fc2_w, fc2_b, out);
}
